// Attention_46566035423948
// MI455X (gfx1250) — matrix operand order repaired and checked by reading, not compiled, not run on hardware
//
#include <hip/hip_runtime.h>
#include <hip/hip_bf16.h>
#include <stdint.h>

// ---------------------------------------------------------------------------
// MI455X (gfx1250) fused attention:
//   k_qkv : X @ W_qkv -> Q,K,V (bf16, [B,H,N,HS]), Q pre-scaled by HS^-0.5
//   k_attn: flash-attention over 32-key chunks; K staged by Tensor Data Mover
//           (double-buffered, TENSORcnt), S/PV on v_wmma_f32_16x16x32_bf16
//   k_proj: AO @ W_proj + b -> fp32 out; A staged by TDM (double-buffered)
// ---------------------------------------------------------------------------

typedef __attribute__((ext_vector_type(16))) __bf16 v16bf;
typedef __attribute__((ext_vector_type(8)))  float  v8f;
typedef __attribute__((ext_vector_type(4)))  int    i32x4;
typedef __attribute__((ext_vector_type(8)))  int    i32x8;

union ABf { v16bf v; unsigned u[8]; };

__device__ __forceinline__ unsigned short f2bf(float f) {
  unsigned u = __float_as_uint(f);
  u += 0x7FFFu + ((u >> 16) & 1u);          // round-to-nearest-even
  return (unsigned short)(u >> 16);
}

// 16-bit A-matrix 16x32 fragment: VGPR i of lane-half h holds K = kbase, kbase+1
__device__ __forceinline__ int a_kbase(int i, int h) {
  return ((i >> 2) << 4) + (h << 3) + ((i & 3) << 1);
}

// Flat shared-aperture addresses keep the LDS byte offset in addr[31:0].
__device__ __forceinline__ unsigned lds_off(const void* p) {
  return (unsigned)(size_t)p;
}

// Tensor Data Mover: load a <=2D bf16 tile (8-byte units) into LDS with
// optional row padding. Group2/3 omitted (2-group form). Issued per-wave,
// tracked by TENSORcnt.
__device__ __forceinline__ void tdm_load_2d(unsigned lds_byte, const void* gaddr,
                                            unsigned tile_d0, unsigned tile_d1,
                                            unsigned tdim0, unsigned tdim1,
                                            unsigned stride0,
                                            unsigned pad_code, int pad_en) {
  unsigned long long ga = (unsigned long long)(size_t)gaddr;
  i32x4 g0;
  g0[0] = 1;                                          // count=1, user mode
  g0[1] = (int)lds_byte;                              // lds_addr
  g0[2] = (int)(unsigned)(ga & 0xFFFFFFFFu);          // global_addr[31:0]
  g0[3] = (int)(((unsigned)(ga >> 32) & 0x01FFFFFFu)  // global_addr[56:32]
                | 0x80000000u);                       // type=2 ("image")
  i32x8 g1;
  g1[0] = (3 << 16) | (pad_en << 20) | ((int)pad_code << 22); // 8B elems, pad_amount=1 DW
  g1[1] = (int)(tdim0 << 16);                         // tensor_dim0 lo16
  g1[2] = (int)((tdim0 >> 16) | (tdim1 << 16));       // tensor_dim0 hi | dim1 lo
  g1[3] = (int)((tdim1 >> 16) | (tile_d0 << 16));     // dim1 hi | tile_dim0
  g1[4] = (int)(tile_d1 & 0xFFFF);                    // tile_dim1 (0 => 1D)
  g1[5] = (int)stride0;                               // tensor_dim0_stride lo32
  g1[6] = 0;
  g1[7] = 0;
  asm volatile("tensor_load_to_lds %0, %1" :: "s"(g0), "s"(g1) : "memory");
}

#define WMMA_BF16(A, B, C) \
  __builtin_amdgcn_wmma_f32_16x16x32_bf16(false, (A), false, (B), (short)0, (C), false, false)

static constexpr int Bsz = 4, Nsq = 2048, Cdim = 1024, Hh = 16, HS = 64;
static constexpr float SCALE = 0.125f;      // 64^-0.5

// ---------------------------------------------------------------------------
// Kernel 1: QKV = X[8192,1024] @ Wqkv[1024,3072]; scatter to Q/K/V bf16
// (both operands need fp32->bf16 conversion, so staging stays manual)
// ---------------------------------------------------------------------------
__global__ __launch_bounds__(128) void k_qkv(const float* __restrict__ X,
                                             const float* __restrict__ W,
                                             unsigned short* __restrict__ Qb,
                                             unsigned short* __restrict__ Kb,
                                             unsigned short* __restrict__ Vb) {
  __shared__ unsigned short Xs[64][34];     // [row][k]  (+2 pad: no bank conflicts)
  __shared__ unsigned short Wt[64][34];     // [col][k]  (W transposed)
  const int t = threadIdx.x, lane = t & 31, wave = t >> 5;
  const int h = lane >> 4, ln = lane & 15;
  const int rowBase = blockIdx.x * 64;
  const int colBase = blockIdx.y * 64;

  v8f acc[4] = {{0,0,0,0,0,0,0,0},{0,0,0,0,0,0,0,0},
                {0,0,0,0,0,0,0,0},{0,0,0,0,0,0,0,0}};

  for (int kc = 0; kc < 1024; kc += 32) {
    if (kc + 32 < 1024) {                   // pull next chunk toward caches
      __builtin_prefetch(&X[(size_t)(rowBase + (t >> 1)) * 1024 + kc + 32 + (t & 1) * 16], 0, 0);
      __builtin_prefetch(&W[(size_t)(kc + 32 + (t >> 2)) * 3072 + colBase + (t & 3) * 16], 0, 0);
    }
#pragma unroll
    for (int r = 0; r < 16; ++r) {          // stage X tile 64x32 fp32->bf16
      int idx = t + r * 128, row = idx >> 5, col = idx & 31;
      Xs[row][col] = f2bf(X[(size_t)(rowBase + row) * 1024 + kc + col]);
    }
#pragma unroll
    for (int r = 0; r < 16; ++r) {          // stage W chunk 32x64, transposed
      int idx = t + r * 128, kr = idx >> 6, col = idx & 63;
      Wt[col][kr] = f2bf(W[(size_t)(kc + kr) * 3072 + colBase + col]);
    }
    __syncthreads();

    ABf a;
#pragma unroll
    for (int i = 0; i < 8; ++i)
      a.u[i] = *(const unsigned*)&Xs[wave * 16 + ln][a_kbase(i, h)];
#pragma unroll
    for (int s = 0; s < 4; ++s) {
      ABf bf;
#pragma unroll
      for (int i = 0; i < 8; ++i)
        bf.u[i] = *(const unsigned*)&Wt[s * 16 + ln][a_kbase(i, h)];
      acc[s] = WMMA_BF16(a.v, bf.v, acc[s]);
    }
    __syncthreads();
  }

  // epilogue: scatter into Q/K/V [B,H,N,HS] (bf16); fold SCALE into Q
#pragma unroll
  for (int s = 0; s < 4; ++s) {
    int gcol = colBase + s * 16 + ln;
    int which = gcol >> 10;                 // 0=Q 1=K 2=V
    int c = gcol & 1023;
    int head = c >> 6, hs = c & 63;
    unsigned short* dst = (which == 0) ? Qb : (which == 1 ? Kb : Vb);
    float scl = (which == 0) ? SCALE : 1.0f;
#pragma unroll
    for (int j = 0; j < 8; ++j) {
      int gm = rowBase + wave * 16 + h * 8 + j;
      int bb = gm >> 11, nn = gm & 2047;
      dst[(((size_t)bb * Hh + head) * Nsq + nn) * HS + hs] = f2bf(acc[s][j] * scl);
    }
  }
}

// ---------------------------------------------------------------------------
// Kernel 2: flash attention. Block = (b,h, 64 q-rows); wave = 16 q-rows.
// K chunks staged by TDM (double-buffered); V transposed manually.
// ---------------------------------------------------------------------------
__global__ __launch_bounds__(128) void k_attn(const unsigned short* __restrict__ Qb,
                                              const unsigned short* __restrict__ Kb,
                                              const unsigned short* __restrict__ Vb,
                                              const int* __restrict__ mask,
                                              unsigned short* __restrict__ AO) {
  __shared__ unsigned short KsB[2][32][66];   // K chunks [key][d], TDM-padded rows
  __shared__ unsigned short Vt[64][34];       // V chunk transposed [d][key]
  __shared__ unsigned short Pt[4][16][34];    // per-wave P tile [qrow][key]
  const int t = threadIdx.x, lane = t & 31, wave = t >> 5;
  const int h = lane >> 4, ln = lane & 15;
  const int q0 = blockIdx.x * 64;
  const int bh = blockIdx.y;                  // b*16 + head
  const int b = bh >> 4, head = bh & 15;
  const size_t base = (size_t)bh * Nsq * HS;
  const unsigned short* Qp = Qb + base;
  const unsigned short* Kp = Kb + base;
  const unsigned short* Vp = Vb + base;
  const int qrow0 = q0 + wave * 16;

  // load this wave's Q A-fragments (2 d-steps of 32) straight from global
  ABf qa[2];
  {
    const unsigned short* qr = Qp + (size_t)(qrow0 + ln) * HS;
#pragma unroll
    for (int s2 = 0; s2 < 2; ++s2)
#pragma unroll
      for (int i = 0; i < 8; ++i)
        qa[s2].u[i] = *(const unsigned*)&qr[s2 * 32 + a_kbase(i, h)];
  }

  // TDM prologue: K chunk 0 = 4KB contiguous; pad 1 DWORD per 32 DWORDs
  if (wave == 0)
    tdm_load_2d(lds_off(&KsB[0][0][0]), Kp, /*tile*/512, 0,
                /*tensor*/512, 1, /*stride*/512, /*pad 32DW*/4, 1);

  float m_run[8], l_run[8];
  v8f acc[4] = {{0,0,0,0,0,0,0,0},{0,0,0,0,0,0,0,0},
                {0,0,0,0,0,0,0,0},{0,0,0,0,0,0,0,0}};
#pragma unroll
  for (int j = 0; j < 8; ++j) { m_run[j] = -3.0e38f; l_run[j] = 0.0f; }

  for (int it = 0; it < Nsq / 32; ++it) {
    const int kc = it * 32;
    const int cb = it & 1;
    // kick off next K chunk while we stage V and compute on this one
    if (wave == 0 && it + 1 < Nsq / 32)
      tdm_load_2d(lds_off(&KsB[cb ^ 1][0][0]),
                  (const unsigned short*)Kp + (size_t)(kc + 32) * HS,
                  512, 0, 512, 1, 512, 4, 1);

    // ---- stage V chunk transposed [64][32] ----
#pragma unroll
    for (int r = 0; r < 8; ++r) {
      int idx = t + r * 128, key = idx >> 5, c2 = idx & 31;
      unsigned vv = *((const unsigned*)Vp + (size_t)(kc + key) * 32 + c2);
      Vt[2 * c2][key]     = (unsigned short)(vv & 0xFFFFu);
      Vt[2 * c2 + 1][key] = (unsigned short)(vv >> 16);
    }
    if (wave == 0) {                         // retire the in-flight K chunk
      if (it + 1 < Nsq / 32) __builtin_amdgcn_s_wait_tensorcnt(1);
      else                   __builtin_amdgcn_s_wait_tensorcnt(0);
    }
    __syncthreads();

    // ---- S = Q @ K^T  (two 16-key subtiles, d = 2x32) ----
    float sv[2][8];
#pragma unroll
    for (int ks = 0; ks < 2; ++ks) {
      v8f s_acc = {0,0,0,0,0,0,0,0};
#pragma unroll
      for (int ds = 0; ds < 2; ++ds) {
        ABf bf;
#pragma unroll
        for (int i = 0; i < 8; ++i)
          bf.u[i] = *(const unsigned*)&KsB[cb][ks * 16 + ln][ds * 32 + a_kbase(i, h)];
        s_acc = WMMA_BF16(qa[ds].v, bf.v, s_acc);
      }
      int key = kc + ks * 16 + ln;
#pragma unroll
      for (int j = 0; j < 8; ++j) {          // mask: where(mask==0, -1e9)
        int q = qrow0 + h * 8 + j;
        int mv = mask[(size_t)q * Nsq + key];
        sv[ks][j] = mv ? s_acc[j] : -1.0e9f;
      }
    }

    // ---- online softmax (rows live across 16 lanes of a half-wave) ----
    float pj[2][8];
#pragma unroll
    for (int j = 0; j < 8; ++j) {
      float mx = fmaxf(sv[0][j], sv[1][j]);
      mx = fmaxf(mx, __shfl_xor(mx, 1, 32));
      mx = fmaxf(mx, __shfl_xor(mx, 2, 32));
      mx = fmaxf(mx, __shfl_xor(mx, 4, 32));
      mx = fmaxf(mx, __shfl_xor(mx, 8, 32));
      float m_new = fmaxf(m_run[j], mx);
      float alpha = __expf(m_run[j] - m_new);
      float p0 = __expf(sv[0][j] - m_new);
      float p1 = __expf(sv[1][j] - m_new);
      float rs = p0 + p1;
      rs += __shfl_xor(rs, 1, 32);
      rs += __shfl_xor(rs, 2, 32);
      rs += __shfl_xor(rs, 4, 32);
      rs += __shfl_xor(rs, 8, 32);
      l_run[j] = l_run[j] * alpha + rs;
      m_run[j] = m_new;
      pj[0][j] = p0; pj[1][j] = p1;
#pragma unroll
      for (int d = 0; d < 4; ++d) acc[d][j] *= alpha;
    }

    // ---- transpose P (C-layout -> A-fragment) through per-wave LDS ----
#pragma unroll
    for (int ks = 0; ks < 2; ++ks)
#pragma unroll
      for (int j = 0; j < 8; ++j)
        Pt[wave][h * 8 + j][ks * 16 + ln] = f2bf(pj[ks][j]);
    ABf pa;
#pragma unroll
    for (int i = 0; i < 8; ++i)
      pa.u[i] = *(const unsigned*)&Pt[wave][ln][a_kbase(i, h)];

    // ---- O += P @ V ----
#pragma unroll
    for (int d = 0; d < 4; ++d) {
      ABf vb;
#pragma unroll
      for (int i = 0; i < 8; ++i)
        vb.u[i] = *(const unsigned*)&Vt[d * 16 + ln][a_kbase(i, h)];
      acc[d] = WMMA_BF16(pa.v, vb.v, acc[d]);
    }
    __syncthreads();
  }

  // ---- finalize: O /= l ; write AO as [B,N,C] bf16 ----
#pragma unroll
  for (int d = 0; d < 4; ++d) {
    int col = head * HS + d * 16 + ln;
#pragma unroll
    for (int j = 0; j < 8; ++j) {
      int row = qrow0 + h * 8 + j;
      AO[((size_t)b * Nsq + row) * Cdim + col] = f2bf(acc[d][j] / l_run[j]);
    }
  }
}

// ---------------------------------------------------------------------------
// Kernel 3: out = AO[8192,1024](bf16) @ Wproj[1024,1024] + b  (fp32 out)
// A chunks staged by TDM 2D tiles (64 rows x 64B, row stride 2KB), padded.
// ---------------------------------------------------------------------------
__global__ __launch_bounds__(128) void k_proj(const unsigned short* __restrict__ A,
                                              const float* __restrict__ W,
                                              const float* __restrict__ bias,
                                              float* __restrict__ out) {
  __shared__ unsigned short XsB[2][64][34];
  __shared__ unsigned short Wt[64][34];
  const int t = threadIdx.x, lane = t & 31, wave = t >> 5;
  const int h = lane >> 4, ln = lane & 15;
  const int rowBase = blockIdx.x * 64;
  const int colBase = blockIdx.y * 64;

  v8f acc[4] = {{0,0,0,0,0,0,0,0},{0,0,0,0,0,0,0,0},
                {0,0,0,0,0,0,0,0},{0,0,0,0,0,0,0,0}};

  // TDM prologue: A chunk 0 (2D: tile_d0=8x8B, 64 rows, stride 256x8B)
  if (wave == 0)
    tdm_load_2d(lds_off(&XsB[0][0][0]), A + (size_t)rowBase * Cdim,
                /*tile*/8, 64, /*tensor*/8, 64, /*stride*/256, /*pad 16DW*/3, 1);

  for (int it = 0; it < 32; ++it) {
    const int kc = it * 32;
    const int cb = it & 1;
    if (wave == 0 && it + 1 < 32)
      tdm_load_2d(lds_off(&XsB[cb ^ 1][0][0]),
                  A + (size_t)rowBase * Cdim + kc + 32,
                  8, 64, 8, 64, 256, 3, 1);
    if (kc + 32 < 1024)
      __builtin_prefetch(&W[(size_t)(kc + 32 + (t >> 2)) * 1024 + colBase + (t & 3) * 16], 0, 0);
#pragma unroll
    for (int r = 0; r < 16; ++r) {           // stage W chunk 32x64, transposed
      int idx = t + r * 128, kr = idx >> 6, col = idx & 63;
      Wt[col][kr] = f2bf(W[(size_t)(kc + kr) * 1024 + colBase + col]);
    }
    if (wave == 0) {
      if (it + 1 < 32) __builtin_amdgcn_s_wait_tensorcnt(1);
      else             __builtin_amdgcn_s_wait_tensorcnt(0);
    }
    __syncthreads();

    ABf a;
#pragma unroll
    for (int i = 0; i < 8; ++i)
      a.u[i] = *(const unsigned*)&XsB[cb][wave * 16 + ln][a_kbase(i, h)];
#pragma unroll
    for (int s = 0; s < 4; ++s) {
      ABf bf;
#pragma unroll
      for (int i = 0; i < 8; ++i)
        bf.u[i] = *(const unsigned*)&Wt[s * 16 + ln][a_kbase(i, h)];
      acc[s] = WMMA_BF16(a.v, bf.v, acc[s]);
    }
    __syncthreads();
  }

#pragma unroll
  for (int s = 0; s < 4; ++s) {
    int gcol = colBase + s * 16 + ln;
    float bv = bias[gcol];
#pragma unroll
    for (int j = 0; j < 8; ++j) {
      int gm = rowBase + wave * 16 + h * 8 + j;
      out[(size_t)gm * 1024 + gcol] = acc[s][j] + bv;
    }
  }
}

// ---------------------------------------------------------------------------
extern "C" void kernel_launch(void* const* d_in, const int* in_sizes, int n_in,
                              void* d_out, int out_size, void* d_ws, size_t ws_size,
                              hipStream_t stream) {
  const float* X     = (const float*)d_in[0];
  const int*   mask  = (const int*)d_in[1];
  const float* Wqkv  = (const float*)d_in[2];
  const float* Wproj = (const float*)d_in[3];
  const float* bproj = (const float*)d_in[4];
  float* out = (float*)d_out;

  const size_t qelems = (size_t)Bsz * Hh * Nsq * HS;   // 8M bf16 each
  unsigned short* Qb = (unsigned short*)d_ws;
  unsigned short* Kb = Qb + qelems;
  unsigned short* Vb = Kb + qelems;
  unsigned short* AO = Vb + qelems;                    // 64 MB total

  k_qkv<<<dim3(128, 48), 128, 0, stream>>>(X, Wqkv, Qb, Kb, Vb);
  k_attn<<<dim3(32, 64), 128, 0, stream>>>(Qb, Kb, Vb, mask, AO);
  k_proj<<<dim3(128, 16), 128, 0, stream>>>(AO, Wproj, bproj, out);
}
